// AttentionModel_33809982554167
// MI455X (gfx1250) — matrix operand order repaired and checked by reading, not compiled, not run on hardware
//
#include <hip/hip_runtime.h>
#include <cstdint>
#include <cstddef>

// ---------------------------------------------------------------------------
// Types for CDNA5 WMMA (gfx1250, wave32)
// ---------------------------------------------------------------------------
typedef __attribute__((ext_vector_type(16))) __bf16 v16bf;
typedef __attribute__((ext_vector_type(8)))  __bf16 v8bf;
typedef __attribute__((ext_vector_type(8)))  float  v8f;

#define WMMA_BF16(a, b, c) \
  __builtin_amdgcn_wmma_f32_16x16x32_bf16(false, (a), false, (b), (short)0, (c), false, false)

static __device__ __forceinline__ v16bf cat8(v8bf lo, v8bf hi) {
  return __builtin_shufflevector(lo, hi, 0,1,2,3,4,5,6,7,8,9,10,11,12,13,14,15);
}

// Model dims
#define TT   512
#define BB   32
#define TB   16384   // T*B
#define QD   128
#define SD   256
#define HH   512
#define NH   8
#define DH   64
#define DKEY 640
#define DVAL 1152
#define PP   1152
#define NEGV (-1.0e32f)

// ---------------------------------------------------------------------------
// f32 -> bf16 convert (weights)
// ---------------------------------------------------------------------------
__global__ void f32_to_bf16_kernel(const float* __restrict__ x, __bf16* __restrict__ y, int n) {
  int i = blockIdx.x * blockDim.x + threadIdx.x;
  int stride = gridDim.x * blockDim.x;
  for (; i < n; i += stride) y[i] = (__bf16)x[i];
}

// ---------------------------------------------------------------------------
// Embedding gather: build query(bf16)[TB,640], value(bf16)[TB,1152],
// and pred tail (cols 512..1151) in fp32 + bf16.
// ---------------------------------------------------------------------------
__global__ __launch_bounds__(256) void embed_kernel(
    const int* __restrict__ ans, const int* __restrict__ cans,
    const int* __restrict__ labels, const float* __restrict__ mask,
    const int* __restrict__ q_ids, const int* __restrict__ sub_ids,
    const float* __restrict__ sub_mask,
    const float* __restrict__ Eq, const float* __restrict__ Es,
    const float* __restrict__ Ea, const float* __restrict__ El,
    __bf16* __restrict__ queryB, __bf16* __restrict__ valueB,
    float* __restrict__ predF, __bf16* __restrict__ predB) {
  int row = blockIdx.x;          // 0..TB-1  (row = t*B + b)
  int c   = threadIdx.x;         // 0..255
  float m   = mask[row];
  int   a   = ans[row] - 1;
  int   ca  = cans[row] - 1;
  int   lb  = labels[row];
  int   qid = q_ids[row];

  float s = 0.f;
#pragma unroll
  for (int j = 0; j < 8; ++j) {
    int   sid = sub_ids[row * 8 + j];
    float sm  = sub_mask[row * 8 + j];
    s += Es[(size_t)sid * SD + c] * sm;
  }
  float cav = Ea[(size_t)ca * SD + c];
  float av  = Ea[(size_t)a * SD + c] * m;
  float lv  = El[(size_t)lb * SD + c] * m;
  float qv  = (c < QD) ? Eq[(size_t)qid * QD + c] : 0.f;

  // query = [questions(128) | subjects(256) | ca(256)]
  __bf16* qr = queryB + (size_t)row * DKEY;
  if (c < QD) qr[c] = (__bf16)qv;
  qr[QD + c]      = (__bf16)s;
  qr[QD + SD + c] = (__bf16)cav;

  // value = [lab(256) | ans(256) | ca(256) | questions(128) | subjects(256)]
  __bf16* vr = valueB + (size_t)row * DVAL;
  vr[c]          = (__bf16)lv;
  vr[SD + c]     = (__bf16)av;
  vr[2 * SD + c] = (__bf16)cav;
  if (c < QD) vr[3 * SD + c] = (__bf16)qv;
  vr[3 * SD + QD + c] = (__bf16)s;

  // pred tail: cols [512..1151] = [questions | subjects | ca]
  float*  pf = predF + (size_t)row * PP;
  __bf16* pb = predB + (size_t)row * PP;
  if (c < QD) { pf[HH + c] = qv; pb[HH + c] = (__bf16)qv; }
  pf[HH + QD + c]      = s;    pb[HH + QD + c]      = (__bf16)s;
  pf[HH + QD + SD + c] = cav;  pb[HH + QD + SD + c] = (__bf16)cav;
}

// ---------------------------------------------------------------------------
// Async-staged bf16 WMMA GEMM:  C[M,N] = act(A[M,K] @ W[N,K]^T + bias)
//
// Block: 256 threads = 8 waves. Block tile 128Mx128N; wave tile 32Mx64N.
// K staged in 64-wide chunks through LDS, double-buffered, filled with
// GLOBAL_LOAD_ASYNC_TO_LDS_B128 (ASYNCcnt) and consumed via ds_load
// fragment reads with contiguous k = 16*half + s for both A and B.
// LDS rows padded to 72 bf16 (144 B) -> 16B-aligned, conflict-spread reads.
// grid = (N/128, M/128). K % 64 == 0.
// ---------------------------------------------------------------------------
#define KC      64
#define LROW    72                       // padded row length in bf16
#define LBUF    (128 * LROW * 2)         // one buffer, bytes (18432)

static __device__ __forceinline__ void stage_chunk_async(
    const __bf16* __restrict__ gA, const __bf16* __restrict__ gW,
    int K, int bm, int bn, int kk, uint32_t ldsA, uint32_t ldsB, int tid) {
#pragma unroll
  for (int t = 0; t < 4; ++t) {
    int sid = tid + t * 256;             // 0..1023
    int row = sid >> 3;                  // 0..127
    int seg = sid & 7;                   // 8 x 16B segments per 64-elem row
    uint32_t goffA = (uint32_t)(((bm + row) * K + kk + seg * 8) * 2);
    uint32_t goffB = (uint32_t)(((bn + row) * K + kk + seg * 8) * 2);
    uint32_t la = ldsA + (uint32_t)(row * (LROW * 2) + seg * 16);
    uint32_t lb = ldsB + (uint32_t)(row * (LROW * 2) + seg * 16);
    asm volatile("global_load_async_to_lds_b128 %0, %1, %2"
                 :: "v"(la), "v"(goffA), "s"(gA) : "memory");
    asm volatile("global_load_async_to_lds_b128 %0, %1, %2"
                 :: "v"(lb), "v"(goffB), "s"(gW) : "memory");
  }
}

template <int ACT, int WF, int WB>
__global__ __launch_bounds__(256) void gemm_bf16_kernel(
    const __bf16* __restrict__ A, const __bf16* __restrict__ W,
    const float* __restrict__ bias,
    float* __restrict__ outF, __bf16* __restrict__ outB,
    int M, int N, int K) {
  __shared__ __align__(16) __bf16 sA[2][128][LROW];
  __shared__ __align__(16) __bf16 sB[2][128][LROW];

  const int tid  = threadIdx.x;
  const int lane = tid & 31;
  const int w    = tid >> 5;
  const int wm   = w & 3;                // 0..3 : 32-row wave strip
  const int wn   = w >> 2;               // 0..1 : 64-col wave strip
  const int bm   = blockIdx.y * 128;
  const int bn   = blockIdx.x * 128;
  const int mrow = lane & 15;
  const int half = lane >> 4;

  // Flat LDS aperture: addr[31:0] is the LDS byte offset (ISA 10.2).
  const uint32_t ldsA0 = (uint32_t)(uintptr_t)&sA[0][0][0];
  const uint32_t ldsB0 = (uint32_t)(uintptr_t)&sB[0][0][0];

  v8f acc[2][4];
#pragma unroll
  for (int s = 0; s < 2; ++s)
#pragma unroll
    for (int j = 0; j < 4; ++j) acc[s][j] = (v8f){};

  stage_chunk_async(A, W, K, bm, bn, 0, ldsA0, ldsB0, tid);

  const int nk = K / KC;
  for (int kc = 0; kc < nk; ++kc) {
    const int cur = kc & 1;
    if (kc + 1 < nk) {
      stage_chunk_async(A, W, K, bm, bn, (kc + 1) * KC,
                        ldsA0 + (uint32_t)((cur ^ 1) * LBUF),
                        ldsB0 + (uint32_t)((cur ^ 1) * LBUF), tid);
      asm volatile("s_wait_asynccnt 8" ::: "memory");   // chunk kc complete
    } else {
      asm volatile("s_wait_asynccnt 0" ::: "memory");
    }
    __syncthreads();

#pragma unroll
    for (int ks = 0; ks < 2; ++ks) {     // two 32-wide k-steps per chunk
      const __bf16* pa0 = &sA[cur][wm * 32 + mrow][ks * 32 + half * 16];
      const __bf16* pa1 = &sA[cur][wm * 32 + 16 + mrow][ks * 32 + half * 16];
      v16bf a0 = cat8(*(const v8bf*)pa0, *(const v8bf*)(pa0 + 8));
      v16bf a1 = cat8(*(const v8bf*)pa1, *(const v8bf*)(pa1 + 8));
#pragma unroll
      for (int j = 0; j < 4; ++j) {
        v16bf bfr = *(const v16bf*)&sB[cur][wn * 64 + j * 16 + mrow][ks * 32 + half * 16];
        acc[0][j] = WMMA_BF16(a0, bfr, acc[0][j]);
        acc[1][j] = WMMA_BF16(a1, bfr, acc[1][j]);
      }
    }
    __syncthreads();
  }

  // Epilogue: C/D layout -> row = base + half*8 + r, col = base + (lane&15)
#pragma unroll
  for (int s = 0; s < 2; ++s) {
    const int row0 = bm + wm * 32 + s * 16 + half * 8;
#pragma unroll
    for (int j = 0; j < 4; ++j) {
      const int col = bn + wn * 64 + j * 16 + mrow;
      const float bv = bias[col];
#pragma unroll
      for (int r = 0; r < 8; ++r) {
        float v = acc[s][j][r] + bv;
        if (ACT) v = fmaxf(v, 0.f);
        size_t idx = (size_t)(row0 + r) * N + col;
        if (WF) outF[idx] = v;
        if (WB) outB[idx] = (__bf16)v;
      }
    }
  }
}

// ---------------------------------------------------------------------------
// V transpose: vT[(b*NH+h)*DH + d][t] = v[(t*B+b)*H + h*DH + d]   (bf16)
// ---------------------------------------------------------------------------
__global__ void transpose_v_kernel(const __bf16* __restrict__ v, __bf16* __restrict__ vT) {
  int i = blockIdx.x * blockDim.x + threadIdx.x;   // 0 .. TB*H-1
  int col = i & (HH - 1);        // h*64 + d
  int row = i >> 9;              // t*B + b
  int t = row >> 5;              // /32
  int b = row & 31;
  int h = col >> 6;
  int d = col & 63;
  vT[((size_t)(b * NH + h) * DH + d) * TT + t] = v[i];
}

// ---------------------------------------------------------------------------
// Flash attention with distance decay + strict causal mask.
// grid = (T/128, B*NH); block = 256 (8 waves). Each wave: 16 queries.
// ---------------------------------------------------------------------------
__global__ __launch_bounds__(256) void attn_kernel(
    const __bf16* __restrict__ qB, const __bf16* __restrict__ kB,
    const __bf16* __restrict__ vT, const float* __restrict__ mask,
    const float* __restrict__ gammas, __bf16* __restrict__ attnB) {
  __shared__ __align__(16) __bf16 plds[8][16][40];   // per-wave P staging (C->A layout)

  const int lane = threadIdx.x & 31;
  const int w    = threadIdx.x >> 5;
  const int bh   = blockIdx.y;
  const int b    = bh >> 3;
  const int h    = bh & 7;
  const int q0   = blockIdx.x * 128 + w * 16;
  const int mrow = lane & 15;
  const int half = lane >> 4;

  const float g  = gammas[h];
  const float sp = log1pf(__expf(g));      // softplus(gamma)
  const float scale = 0.125f;              // 1/sqrt(64)

  // Q fragments over dh (two 16x32 frags)
  const __bf16* qrow = qB + ((size_t)(q0 + mrow) * BB + b) * HH + h * DH + half * 16;
  v16bf aq0 = cat8(*(const v8bf*)(qrow),      *(const v8bf*)(qrow + 8));
  v16bf aq1 = cat8(*(const v8bf*)(qrow + 32), *(const v8bf*)(qrow + 40));

  float mrun[8], lrun[8];
#pragma unroll
  for (int r = 0; r < 8; ++r) { mrun[r] = -3.0e38f; lrun[r] = 0.f; }
  v8f o0 = {}, o1 = {}, o2 = {}, o3 = {};

  const int kend  = q0 + 16;               // strict causal: keys < max query
  const int nIter = (kend + 31) >> 5;

  for (int it = 0; it < nIter; ++it) {
    const int kb = it * 32;
    // ---- S = Q K^T (two 16-key sub-tiles) ----
    v8f s0 = {}, s1 = {};
    {
      const __bf16* kr0 = kB + ((size_t)((kb + mrow) * BB) + b) * HH + h * DH + half * 16;
      v16bf b00 = *(const v16bf*)(kr0);
      v16bf b01 = *(const v16bf*)(kr0 + 32);
      s0 = WMMA_BF16(aq0, b00, s0);
      s0 = WMMA_BF16(aq1, b01, s0);
      const __bf16* kr1 = kr0 + (size_t)16 * BB * HH;
      v16bf b10 = *(const v16bf*)(kr1);
      v16bf b11 = *(const v16bf*)(kr1 + 32);
      s1 = WMMA_BF16(aq0, b10, s1);
      s1 = WMMA_BF16(aq1, b11, s1);
    }
    // ---- masks, decay, online softmax ----
    const int k0i = kb + (lane & 15);
    const int k1i = k0i + 16;
    const float mk0 = mask[k0i * BB + b];
    const float mk1 = mask[k1i * BB + b];
#pragma unroll
    for (int r = 0; r < 8; ++r) {
      const int qi = q0 + half * 8 + r;
      float v0 = s0[r] * scale;
      float v1 = s1[r] * scale;
      if (mk0 == 0.f) v0 = NEGV;
      if (mk1 == 0.f) v1 = NEGV;
      float te0 = __expf(-sp * sqrtf(fabsf((float)(qi - k0i))));
      float te1 = __expf(-sp * sqrtf(fabsf((float)(qi - k1i))));
      te0 = fminf(fmaxf(te0, 1e-5f), 1e5f);
      te1 = fminf(fmaxf(te1, 1e-5f), 1e5f);
      v0 *= te0;
      v1 *= te1;
      if (k0i >= qi) v0 = NEGV;
      if (k1i >= qi) v1 = NEGV;

      float tm = fmaxf(v0, v1);
#pragma unroll
      for (int off = 1; off < 16; off <<= 1) tm = fmaxf(tm, __shfl_xor(tm, off));
      const float newm  = fmaxf(mrun[r], tm);
      const float alpha = __expf(mrun[r] - newm);
      const float p0 = __expf(v0 - newm);
      const float p1 = __expf(v1 - newm);
      float rs = p0 + p1;
#pragma unroll
      for (int off = 1; off < 16; off <<= 1) rs += __shfl_xor(rs, off);
      lrun[r] = lrun[r] * alpha + rs;
      mrun[r] = newm;
      o0[r] *= alpha; o1[r] *= alpha; o2[r] *= alpha; o3[r] *= alpha;

      plds[w][half * 8 + r][lane & 15]        = (__bf16)p0;
      plds[w][half * 8 + r][16 + (lane & 15)] = (__bf16)p1;
    }
    asm volatile("s_wait_dscnt 0" ::: "memory");
    // ---- reload P as 16x32 A fragment ----
    const __bf16* pr = &plds[w][mrow][0];
    v16bf pa = cat8(*(const v8bf*)(pr + half * 16), *(const v8bf*)(pr + half * 16 + 8));
    // ---- O += P @ V  (B frags from pre-transposed V^T: contiguous keys) ----
    const __bf16* vtp = vT + ((size_t)bh * DH + (lane & 15)) * TT + kb + half * 16;
    o0 = WMMA_BF16(pa, *(const v16bf*)(vtp),              o0);
    o1 = WMMA_BF16(pa, *(const v16bf*)(vtp + 16 * TT),    o1);
    o2 = WMMA_BF16(pa, *(const v16bf*)(vtp + 32 * TT),    o2);
    o3 = WMMA_BF16(pa, *(const v16bf*)(vtp + 48 * TT),    o3);
  }

  // ---- normalize + store (query 0 row is zeroed per reference) ----
#pragma unroll
  for (int r = 0; r < 8; ++r) {
    const int ti = q0 + half * 8 + r;
    float inv = (lrun[r] > 0.f) ? 1.0f / lrun[r] : 0.f;
    if (ti == 0) inv = 0.f;
    size_t base = ((size_t)ti * BB + b) * HH + h * DH + (lane & 15);
    attnB[base]      = (__bf16)(o0[r] * inv);
    attnB[base + 16] = (__bf16)(o1[r] * inv);
    attnB[base + 32] = (__bf16)(o2[r] * inv);
    attnB[base + 48] = (__bf16)(o3[r] * inv);
  }
}

// ---------------------------------------------------------------------------
// LayerNorm over H=512 (one row per block), optional residual add.
// ---------------------------------------------------------------------------
__global__ __launch_bounds__(256) void layernorm512_kernel(
    const float* __restrict__ x, const float* __restrict__ g,
    const float* __restrict__ bb, const float* __restrict__ res,
    float* __restrict__ outF, long ldF, __bf16* __restrict__ outB, long ldB) {
  __shared__ float red[256];
  const int row = blockIdx.x;
  const int tid = threadIdx.x;
  const float* xr = x + (size_t)row * HH;
  const float a0 = xr[tid], a1 = xr[tid + 256];

  red[tid] = a0 + a1;
  __syncthreads();
  for (int o = 128; o > 0; o >>= 1) { if (tid < o) red[tid] += red[tid + o]; __syncthreads(); }
  const float mu = red[0] * (1.0f / HH);
  __syncthreads();
  const float d0 = a0 - mu, d1 = a1 - mu;
  red[tid] = d0 * d0 + d1 * d1;
  __syncthreads();
  for (int o = 128; o > 0; o >>= 1) { if (tid < o) red[tid] += red[tid + o]; __syncthreads(); }
  const float var  = red[0] * (1.0f / HH);
  const float rinv = rsqrtf(var + 1e-5f);

  float v0 = d0 * rinv * g[tid]       + bb[tid];
  float v1 = d1 * rinv * g[tid + 256] + bb[tid + 256];
  if (res) {
    v0 += res[(size_t)row * HH + tid];
    v1 += res[(size_t)row * HH + tid + 256];
  }
  if (outF) { outF[(size_t)row * ldF + tid] = v0; outF[(size_t)row * ldF + tid + 256] = v1; }
  if (outB) { outB[(size_t)row * ldB + tid] = (__bf16)v0; outB[(size_t)row * ldB + tid + 256] = (__bf16)v1; }
}

// ---------------------------------------------------------------------------
// Final logits + BCE loss partial sums. One wave per row (8 rows per block).
// ---------------------------------------------------------------------------
__global__ __launch_bounds__(256) void loss_kernel(
    const float* __restrict__ h2, const float* __restrict__ predF,
    const float* __restrict__ OW, const float* __restrict__ Ob,
    const int* __restrict__ labels, const float* __restrict__ mask,
    float* __restrict__ acc) {
  __shared__ float sb[8], sm[8];
  const int lane = threadIdx.x & 31;
  const int w    = threadIdx.x >> 5;
  const int row  = blockIdx.x * 8 + w;
  const float* hr = h2    + (size_t)row * PP;
  const float* pr = predF + (size_t)row * PP;
  float dot = 0.f;
  for (int c = lane; c < PP; c += 32) dot += (hr[c] + pr[c]) * OW[c];
#pragma unroll
  for (int off = 16; off > 0; off >>= 1) dot += __shfl_xor(dot, off);
  if (lane == 0) {
    const float logit = dot + Ob[0];
    const float y = (float)labels[row];
    const float m = mask[row];
    const float bce = fmaxf(logit, 0.f) - logit * y + log1pf(__expf(-fabsf(logit)));
    sb[w] = bce * m;
    sm[w] = m;
  }
  __syncthreads();
  if (threadIdx.x == 0) {
    float s1 = 0.f, s2 = 0.f;
#pragma unroll
    for (int i = 0; i < 8; ++i) { s1 += sb[i]; s2 += sm[i]; }
    atomicAdd(&acc[0], s1);
    atomicAdd(&acc[1], s2);
  }
}

__global__ void init_acc_kernel(float* acc) { if (threadIdx.x < 2) acc[threadIdx.x] = 0.f; }
__global__ void finalize_kernel(const float* acc, float* out) { out[0] = acc[0] / acc[1]; }

// ---------------------------------------------------------------------------
// Host launch
// ---------------------------------------------------------------------------
extern "C" void kernel_launch(void* const* d_in, const int* in_sizes, int n_in,
                              void* d_out, int out_size, void* d_ws, size_t ws_size,
                              hipStream_t stream) {
  (void)in_sizes; (void)n_in; (void)out_size; (void)ws_size;
  const int*   ans      = (const int*)d_in[0];
  const int*   cans     = (const int*)d_in[1];
  const int*   labels   = (const int*)d_in[2];
  const float* mask     = (const float*)d_in[3];
  const int*   q_ids    = (const int*)d_in[4];
  const int*   sub_ids  = (const int*)d_in[5];
  const float* sub_mask = (const float*)d_in[6];
  const float* Eq  = (const float*)d_in[7];
  const float* Es  = (const float*)d_in[8];
  const float* Ea  = (const float*)d_in[9];
  const float* El  = (const float*)d_in[10];
  const float* Wq  = (const float*)d_in[11]; const float* bq  = (const float*)d_in[12];
  const float* Wk  = (const float*)d_in[13]; const float* bk  = (const float*)d_in[14];
  const float* Wv  = (const float*)d_in[15]; const float* bv  = (const float*)d_in[16];
  const float* Wo  = (const float*)d_in[17]; const float* bo  = (const float*)d_in[18];
  const float* ln1g = (const float*)d_in[19]; const float* ln1b = (const float*)d_in[20];
  const float* W1  = (const float*)d_in[21]; const float* b1  = (const float*)d_in[22];
  const float* W2  = (const float*)d_in[23]; const float* b2  = (const float*)d_in[24];
  const float* ln2g = (const float*)d_in[25]; const float* ln2b = (const float*)d_in[26];
  const float* gammas = (const float*)d_in[27];
  const float* L1W = (const float*)d_in[28]; const float* L1b = (const float*)d_in[29];
  const float* L2W = (const float*)d_in[30]; const float* L2b = (const float*)d_in[31];
  const float* OW  = (const float*)d_in[32]; const float* Ob  = (const float*)d_in[33];
  float* out = (float*)d_out;

  // ---- workspace layout (explicit slot reuse; peak ~311 MB) ----
  char* base = (char*)d_ws;
  size_t off = 0;
  auto alloc = [&](size_t bytes) { size_t o = off; off += (bytes + 255) & ~(size_t)255; return o; };

  const size_t SLOT = (size_t)TB * PP * 2;   // 37,748,736 B

  size_t o_wq = alloc((size_t)HH * DKEY * 2);
  size_t o_wk = alloc((size_t)HH * DKEY * 2);
  size_t o_wv = alloc((size_t)HH * DVAL * 2);
  size_t o_wo = alloc((size_t)HH * HH * 2);
  size_t o_w1 = alloc((size_t)HH * HH * 2);
  size_t o_w2 = alloc((size_t)HH * HH * 2);
  size_t o_l1 = alloc((size_t)PP * PP * 2);
  size_t o_l2 = alloc((size_t)PP * PP * 2);
  size_t o_acc   = alloc(256);
  size_t o_predF = alloc((size_t)TB * PP * 4);
  size_t o_predB = alloc((size_t)TB * PP * 2);
  size_t o_s1 = alloc(SLOT);
  size_t o_s2 = alloc(SLOT);
  size_t o_s3 = alloc(SLOT);
  size_t o_s4 = alloc(SLOT);
  size_t o_s5 = alloc(SLOT);

  __bf16* wq_b = (__bf16*)(base + o_wq);
  __bf16* wk_b = (__bf16*)(base + o_wk);
  __bf16* wv_b = (__bf16*)(base + o_wv);
  __bf16* wo_b = (__bf16*)(base + o_wo);
  __bf16* w1_b = (__bf16*)(base + o_w1);
  __bf16* w2_b = (__bf16*)(base + o_w2);
  __bf16* l1_b = (__bf16*)(base + o_l1);
  __bf16* l2_b = (__bf16*)(base + o_l2);
  float*  acc  = (float*)(base + o_acc);
  float*  predF = (float*)(base + o_predF);
  __bf16* predB = (__bf16*)(base + o_predB);

  __bf16* queryB = (__bf16*)(base + o_s1);                       // S1
  __bf16* valueB = (__bf16*)(base + o_s2);                       // S2
  __bf16* q_b    = (__bf16*)(base + o_s3);                       // S3
  __bf16* k_b    = (__bf16*)(base + o_s4);                       // S4
  __bf16* v_b    = (__bf16*)(base + o_s5);                       // S5 (lo)
  __bf16* vT_b   = (__bf16*)(base + o_s5 + (size_t)TB * HH * 2); // S5 (hi)
  __bf16* attnB  = (__bf16*)(base + o_s1);                       // S1 (query dead)
  float*  o_f32  = (float*)(base + o_s2);                        // S2 (value dead)
  float*  out_f32 = (float*)(base + o_s3);                       // S3 (q dead)
  __bf16* out_bf  = (__bf16*)(base + o_s4);                      // S4 (k dead)
  __bf16* ffn1_b  = (__bf16*)(base + o_s5);                      // S5 (v/vT dead)
  float*  ffn2_f  = (float*)(base + o_s1);                       // S1 (attn dead)
  __bf16* h1_b    = (__bf16*)(base + o_s2);                      // S2 (o_f32 dead)
  float*  h2_f    = (float*)(base + o_s3);                       // S3+S4 (75.5 MB)

  // ---- pipeline ----
  init_acc_kernel<<<1, 32, 0, stream>>>(acc);

  f32_to_bf16_kernel<<<512, 256, 0, stream>>>(Wq, wq_b, HH * DKEY);
  f32_to_bf16_kernel<<<512, 256, 0, stream>>>(Wk, wk_b, HH * DKEY);
  f32_to_bf16_kernel<<<512, 256, 0, stream>>>(Wv, wv_b, HH * DVAL);
  f32_to_bf16_kernel<<<512, 256, 0, stream>>>(Wo, wo_b, HH * HH);
  f32_to_bf16_kernel<<<512, 256, 0, stream>>>(W1, w1_b, HH * HH);
  f32_to_bf16_kernel<<<512, 256, 0, stream>>>(W2, w2_b, HH * HH);
  f32_to_bf16_kernel<<<1024, 256, 0, stream>>>(L1W, l1_b, PP * PP);
  f32_to_bf16_kernel<<<1024, 256, 0, stream>>>(L2W, l2_b, PP * PP);

  embed_kernel<<<TB, 256, 0, stream>>>(ans, cans, labels, mask, q_ids, sub_ids, sub_mask,
                                       Eq, Es, Ea, El, queryB, valueB, predF, predB);

  // q/k/v projections (bf16 out)
  gemm_bf16_kernel<0, 0, 1><<<dim3(HH / 128, TB / 128), 256, 0, stream>>>(
      queryB, wq_b, bq, nullptr, q_b, TB, HH, DKEY);
  gemm_bf16_kernel<0, 0, 1><<<dim3(HH / 128, TB / 128), 256, 0, stream>>>(
      queryB, wk_b, bk, nullptr, k_b, TB, HH, DKEY);
  gemm_bf16_kernel<0, 0, 1><<<dim3(HH / 128, TB / 128), 256, 0, stream>>>(
      valueB, wv_b, bv, nullptr, v_b, TB, HH, DVAL);

  transpose_v_kernel<<<(TB * HH) / 256, 256, 0, stream>>>(v_b, vT_b);

  attn_kernel<<<dim3(TT / 128, BB * NH), 256, 0, stream>>>(q_b, k_b, vT_b, mask, gammas, attnB);

  // Wo projection (f32 out) -> LN1
  gemm_bf16_kernel<0, 1, 0><<<dim3(HH / 128, TB / 128), 256, 0, stream>>>(
      attnB, wo_b, bo, o_f32, nullptr, TB, HH, HH);
  layernorm512_kernel<<<TB, 256, 0, stream>>>(o_f32, ln1g, ln1b, nullptr,
                                              out_f32, HH, out_bf, HH);

  // FFN
  gemm_bf16_kernel<1, 0, 1><<<dim3(HH / 128, TB / 128), 256, 0, stream>>>(
      out_bf, w1_b, b1, nullptr, ffn1_b, TB, HH, HH);
  gemm_bf16_kernel<0, 1, 0><<<dim3(HH / 128, TB / 128), 256, 0, stream>>>(
      ffn1_b, w2_b, b2, ffn2_f, nullptr, TB, HH, HH);
  // out + LN(ffn) -> pred cols [0..511]
  layernorm512_kernel<<<TB, 256, 0, stream>>>(ffn2_f, ln2g, ln2b, out_f32,
                                              predF, PP, predB, PP);

  // prediction head
  gemm_bf16_kernel<1, 0, 1><<<dim3(PP / 128, TB / 128), 256, 0, stream>>>(
      predB, l1_b, L1b, nullptr, h1_b, TB, PP, PP);
  gemm_bf16_kernel<1, 1, 0><<<dim3(PP / 128, TB / 128), 256, 0, stream>>>(
      h1_b, l2_b, L2b, h2_f, nullptr, TB, PP, PP);

  loss_kernel<<<TB / 8, 256, 0, stream>>>(h2_f, predF, OW, Ob, labels, mask, acc);
  finalize_kernel<<<1, 1, 0, stream>>>(acc, out);
}
